// TFAttention_1486058684641
// MI455X (gfx1250) — compile-verified, matrix operand order checked by reading, not run on hardware
//
#include <hip/hip_runtime.h>

// ---------------------------------------------------------------------------
// CDNA5 (gfx1250) fused TFAttention:
//   qkv = x@W_attn+b  -> Q scratch + present text K/V rows
//   vis/tags projections -> present prefix K/V rows
//   flash-style attention (online softmax, async double-buffered K/V) -> a_tmp
//   proj = a_tmp@W_proj+b -> a
// Matrix math: v_wmma_f32_16x16x32_bf16 (f32 -> bf16 operands, f32 accumulate).
// Data movement: global_load_async_to_lds_b128 (ASYNCcnt) for the K/V stream.
// ---------------------------------------------------------------------------

typedef __attribute__((ext_vector_type(16))) __bf16 v16bf;
typedef __attribute__((ext_vector_type(2)))  __bf16 bf16x2;
typedef __attribute__((ext_vector_type(8)))  float  v8f;

#define NH      16
#define DH      64
#define SEQ     1024
#define NX      1024
#define NSK     1087      // 14 tags + 49 visual + 1024 text
#define PREFIX  63        // tags + visual rows (always visible)
#define NKB     34        // ceil(NSK / 32)

// k-offset pattern shared by the 16-bit A fragment (16x32) and our transposed
// B staging (ISA 7.12.2): lanes 0-15 use k-half 0, lanes 16-31 use k-half 8.
__device__ __forceinline__ void frag_ko(int lane, int ko[8]) {
  const int khalf = (lane & 16) ? 8 : 0;
#pragma unroll
  for (int g = 0; g < 4; ++g) { ko[g] = khalf + 2 * g; ko[4 + g] = 16 + khalf + 2 * g; }
}

__device__ __forceinline__ v16bf load_frag(const __bf16* rowptr, int kbase, const int ko[8]) {
  v16bf f;
#pragma unroll
  for (int g = 0; g < 8; ++g) {
    bf16x2 p = *reinterpret_cast<const bf16x2*>(rowptr + kbase + ko[g]);
    f[2 * g + 0] = p[0];
    f[2 * g + 1] = p[1];
  }
  return f;
}

__device__ __forceinline__ v8f wmma_bf16(v16bf a, v16bf b, v8f c) {
  return __builtin_amdgcn_wmma_f32_16x16x32_bf16(false, a, false, b, (short)0, c,
                                                 false, false);
}

__device__ __forceinline__ int imin(int a, int b) { return a < b ? a : b; }

// 32-bit wave-relative LDS address for async-DMA destinations: the low 32 bits
// of a flat shared-aperture address are the LDS byte address (ISA 10.2).
__device__ __forceinline__ unsigned lds_addr32(const void* p) {
  return (unsigned)(unsigned long long)(uintptr_t)p;
}

__device__ __forceinline__ void async_load_b128(unsigned lds_dst, const void* gsrc) {
  asm volatile("global_load_async_to_lds_b128 %0, %1, off"
               :: "v"(lds_dst), "v"((unsigned long long)(uintptr_t)gsrc)
               : "memory");
}

__device__ __forceinline__ void wait_async0() {
  asm volatile("s_wait_asynccnt 0x0" ::: "memory");
}

// ---------------------------------------------------------------------------
// Generic GEMM: C[M,N] = A[M,K] @ B[K,N] + bias[N]
// 128 threads = 4 waves; block tile 64x64; each wave a 32x32 tile (2x2 WMMA).
// MODE 0: plain row-major out
// MODE 1: qkv split (cols 0..1023 -> Q scratch, 1024.. -> present text rows)
// MODE 2: kv prefix (vis/tags) -> present rows [rowOffset, rowOffset+rpb)
// EDGE=false: M%64==0, N%64==0, K%32==0 guaranteed -> branch-free hot loop.
// ---------------------------------------------------------------------------
template <int MODE, bool EDGE>
__global__ __launch_bounds__(128) void gemm_bf16_wmma(
    const float* __restrict__ A, int lda,
    const float* __restrict__ Bw, int ldb,
    const float* __restrict__ bias,
    float* __restrict__ outPlain,
    float* __restrict__ present,
    int M, int N, int K,
    int rowOffset, int rowsPerBatch)
{
  __shared__ __bf16 As[64 * 32];   // [m][k]
  __shared__ __bf16 Bs[64 * 32];   // [n][k] (transposed at stage time)

  const int tid  = threadIdx.x;
  const int lane = tid & 31, wave = tid >> 5;
  const int wr = wave >> 1, wc = wave & 1;
  const int m0 = blockIdx.y * 64, n0 = blockIdx.x * 64;
  int ko[8]; frag_ko(lane, ko);

  const v8f vzero = {0.f, 0.f, 0.f, 0.f, 0.f, 0.f, 0.f, 0.f};
  v8f acc[2][2];
#pragma unroll
  for (int i = 0; i < 2; ++i)
#pragma unroll
    for (int j = 0; j < 2; ++j) acc[i][j] = vzero;

  const int KT  = (K + 31) >> 5;
  const int ar  = tid >> 1, ac0 = (tid & 1) * 16;   // A stage: 1 row, 16 cols
  const int brk = tid >> 2, bc0 = (tid & 3) * 16;   // B stage: 1 k-row, 16 cols

#pragma unroll 1
  for (int kt = 0; kt < KT; ++kt) {
    const int k0 = kt << 5;
    // ---- stage A tile (64x32, f32 -> bf16, packed dword LDS stores) ----
    {
      const int gm = m0 + ar;
      const int gmc = EDGE ? imin(gm, M - 1) : gm;
#pragma unroll
      for (int q = 0; q < 4; ++q) {
        const int c = ac0 + q * 4;
        const int gk = k0 + c;
        float va[4];
        if (!EDGE) {
          const float4 t = *reinterpret_cast<const float4*>(A + (size_t)gm * lda + gk);
          va[0] = t.x; va[1] = t.y; va[2] = t.z; va[3] = t.w;
        } else {
#pragma unroll
          for (int e = 0; e < 4; ++e) {
            const int gke = imin(gk + e, K - 1);
            const float t = A[(size_t)gmc * lda + gke];     // clamped, never faults
            va[e] = (gm < M && gk + e < K) ? t : 0.f;        // v_cndmask, no branch
          }
        }
#pragma unroll
        for (int e = 0; e < 2; ++e) {
          bf16x2 p; p[0] = (__bf16)va[2 * e]; p[1] = (__bf16)va[2 * e + 1];
          *reinterpret_cast<bf16x2*>(&As[ar * 32 + c + 2 * e]) = p;
        }
      }
      if (!EDGE && kt + 1 < KT)
        __builtin_prefetch(A + (size_t)gm * lda + k0 + 32, 0, 0);
    }
    // ---- stage B tile (32x64 -> transposed [n][k], f32 -> bf16) ----
    {
      const int gk = k0 + brk;
      const int gkc = EDGE ? imin(gk, K - 1) : gk;
#pragma unroll
      for (int q = 0; q < 4; ++q) {
        const int c = bc0 + q * 4;
        const int gn = n0 + c;
        float vb[4];
        if (!EDGE) {
          const float4 t = *reinterpret_cast<const float4*>(Bw + (size_t)gk * ldb + gn);
          vb[0] = t.x; vb[1] = t.y; vb[2] = t.z; vb[3] = t.w;
        } else {
#pragma unroll
          for (int e = 0; e < 4; ++e) {
            const int gne = imin(gn + e, N - 1);
            const float t = Bw[(size_t)gkc * ldb + gne];
            vb[e] = (gk < K && gn + e < N) ? t : 0.f;
          }
        }
#pragma unroll
        for (int e = 0; e < 4; ++e) Bs[(c + e) * 32 + brk] = (__bf16)vb[e];
      }
      if (!EDGE && kt + 1 < KT)
        __builtin_prefetch(Bw + (size_t)(gk + 32) * ldb + n0 + bc0, 0, 0);
    }
    __syncthreads();

    v16bf af[2], bfv[2];
#pragma unroll
    for (int t2 = 0; t2 < 2; ++t2) {
      af[t2]  = load_frag(As + (wr * 32 + t2 * 16 + (lane & 15)) * 32, 0, ko);
      bfv[t2] = load_frag(Bs + (wc * 32 + t2 * 16 + (lane & 15)) * 32, 0, ko);
    }
#pragma unroll
    for (int i = 0; i < 2; ++i)
#pragma unroll
      for (int j = 0; j < 2; ++j)
        acc[i][j] = wmma_bf16(af[i], bfv[j], acc[i][j]);
    __syncthreads();
  }

  // ---- epilogue: C-frag element (vgpr r, lane L) = (M = r + 8*(L>>4), N = L&15)
  const int hi = lane >> 4, nl = lane & 15;
#pragma unroll
  for (int i = 0; i < 2; ++i) {
#pragma unroll
    for (int j = 0; j < 2; ++j) {
#pragma unroll
      for (int r = 0; r < 8; ++r) {
        const int gm = m0 + wr * 32 + i * 16 + r + 8 * hi;
        const int gn = n0 + wc * 32 + j * 16 + nl;
        if (EDGE && (gm >= M || gn >= N)) continue;
        const float v = acc[i][j][r] + bias[gn];
        if (MODE == 0) {
          outPlain[(size_t)gm * N + gn] = v;
        } else if (MODE == 1) {
          if (gn < NX) {
            outPlain[(size_t)gm * NX + gn] = v;                   // Q scratch
          } else {
            const int kv = (gn >= 2 * NX) ? 1 : 0;
            const int nn = gn - NX - kv * NX;
            const int hh = nn >> 6, d = nn & 63;
            const int b = gm >> 10, s = gm & 1023;
            present[((((size_t)b * 2 + kv) * NH + hh) * NSK + (PREFIX + s)) * DH + d] = v;
          }
        } else {
          const int kv = (gn >= NX) ? 1 : 0;
          const int nn = gn - kv * NX;
          const int hh = nn >> 6, d = nn & 63;
          const int b = gm / rowsPerBatch, ii = gm % rowsPerBatch;
          present[((((size_t)b * 2 + kv) * NH + hh) * NSK + (rowOffset + ii)) * DH + d] = v;
        }
      }
    }
  }
}

// ---------------------------------------------------------------------------
// Fused attention: one workgroup per (b, h, 64 q-rows); 4 waves x 16 q-rows.
// K/V blocks (32 keys) are DMA'd with global_load_async_to_lds_b128 into a
// double-buffered f32 staging area (copy of block kb+1 overlaps compute of kb),
// then converted once to bf16 tiles for the WMMA fragment loads.
// ---------------------------------------------------------------------------
__global__ __launch_bounds__(128) void attn_fused(
    const float* __restrict__ qtmp,       // [B,S,nx] (q section of qkv)
    const float* __restrict__ present,    // [B,2,H,ns,dh]
    float* __restrict__ a_tmp)            // [B,S,nx] merged-head output
{
  __shared__ float  Kf32[2][32 * 64];   // async staging (double buffer)
  __shared__ float  Vf32[2][32 * 64];
  __shared__ __bf16 Ks[32 * 64];        // [key][d]   (score B-frag: pairs in d)
  __shared__ __bf16 Vs[64 * 32];        // [d][key]   (PV    B-frag: pairs in key)
  __shared__ __bf16 Ps[4 * 16 * 32];    // per-wave probs (C-layout -> A-layout)

  const int tid = threadIdx.x, lane = tid & 31, wave = tid >> 5;
  const int b = blockIdx.z, h = blockIdx.y;
  const int qbase = blockIdx.x * 64 + wave * 16;
  const float* Kg = present + (((size_t)b * 2 + 0) * NH + h) * NSK * DH;
  const float* Vg = present + (((size_t)b * 2 + 1) * NH + h) * NSK * DH;
  int ko[8]; frag_ko(lane, ko);
  const int hi = lane >> 4, nl = lane & 15;
  const int jl = tid >> 2, c0 = (tid & 3) * 16;   // staging: row jl, 16 cols

  // Q fragments for both k-steps (d 0..31 / 32..63); 1/sqrt(dh)=0.125 folded in.
  v16bf qa[2];
  {
    const float* qp = qtmp + ((size_t)b * SEQ + (qbase + nl)) * NX + h * DH;
#pragma unroll
    for (int ks = 0; ks < 2; ++ks) {
#pragma unroll
      for (int g = 0; g < 8; ++g) {
        const float2 t = *reinterpret_cast<const float2*>(qp + ks * 32 + ko[g]);
        qa[ks][2 * g + 0] = (__bf16)(t.x * 0.125f);
        qa[ks][2 * g + 1] = (__bf16)(t.y * 0.125f);
      }
    }
  }

  // issue the async DMA for one 32-key block into staging buffer `buf`
  auto issue_async = [&](int kb2, int buf) {
    const int j = kb2 * 32 + jl;
    if (j < NSK) {                               // pad rows keep stale finite data
#pragma unroll
      for (int q = 0; q < 4; ++q) {
        const int c = c0 + q * 4;
        async_load_b128(lds_addr32(&Kf32[buf][jl * 64 + c]), Kg + (size_t)j * DH + c);
        async_load_b128(lds_addr32(&Vf32[buf][jl * 64 + c]), Vg + (size_t)j * DH + c);
      }
    }
  };

  const v8f vzero = {0.f, 0.f, 0.f, 0.f, 0.f, 0.f, 0.f, 0.f};
  v8f acc[4] = {vzero, vzero, vzero, vzero};
  float mrow[8], lrow[8];
#pragma unroll
  for (int r = 0; r < 8; ++r) { mrow[r] = -3.0e38f; lrow[r] = 0.f; }

  issue_async(0, 0);

#pragma unroll 1
  for (int kb = 0; kb < NKB; ++kb) {
    const int cur = kb & 1;
    const int jb = kb * 32;
    wait_async0();                 // this wave's slice of buffer `cur` landed
    __syncthreads();               // everyone's slice landed
    if (kb + 1 < NKB) issue_async(kb + 1, cur ^ 1);   // overlaps convert+compute

    // ---- convert f32 staging -> bf16 tiles (K packed dwords, V transposed) ----
#pragma unroll
    for (int q = 0; q < 4; ++q) {
      const int c = c0 + q * 4;
      const float4 tk = *reinterpret_cast<const float4*>(&Kf32[cur][jl * 64 + c]);
      const float4 tv = *reinterpret_cast<const float4*>(&Vf32[cur][jl * 64 + c]);
      bf16x2 p0; p0[0] = (__bf16)tk.x; p0[1] = (__bf16)tk.y;
      bf16x2 p1; p1[0] = (__bf16)tk.z; p1[1] = (__bf16)tk.w;
      *reinterpret_cast<bf16x2*>(&Ks[jl * 64 + c + 0]) = p0;
      *reinterpret_cast<bf16x2*>(&Ks[jl * 64 + c + 2]) = p1;
      Vs[(c + 0) * 32 + jl] = (__bf16)tv.x;
      Vs[(c + 1) * 32 + jl] = (__bf16)tv.y;
      Vs[(c + 2) * 32 + jl] = (__bf16)tv.z;
      Vs[(c + 3) * 32 + jl] = (__bf16)tv.w;
    }
    __syncthreads();

    // ---- scores S = Q.K^T for two 16-key subtiles ----
    v8f sc[2];
#pragma unroll
    for (int nsub = 0; nsub < 2; ++nsub) {
      const __bf16* krow = Ks + (nsub * 16 + nl) * 64;
      const v16bf kf0 = load_frag(krow, 0, ko);    // d 0..31
      const v16bf kf1 = load_frag(krow, 32, ko);   // d 32..63
      v8f sv = wmma_bf16(qa[0], kf0, vzero);
      sc[nsub] = wmma_bf16(qa[1], kf1, sv);
    }

    // ---- mask + online softmax (row stats uniform per 16-lane half) ----
    const int j0 = jb + nl, j1 = j0 + 16;
#pragma unroll
    for (int r = 0; r < 8; ++r) {
      const int irow = qbase + r + 8 * hi;
      float x0 = sc[0][r], x1 = sc[1][r];
      if (j0 >= NSK) x0 = -3.0e38f;
      else if (j0 >= PREFIX && (j0 - PREFIX) > irow) x0 = -10000.f;
      if (j1 >= NSK) x1 = -3.0e38f;
      else if (j1 >= PREFIX && (j1 - PREFIX) > irow) x1 = -10000.f;

      float mx = fmaxf(x0, x1);
#pragma unroll
      for (int msk = 1; msk <= 8; msk <<= 1) mx = fmaxf(mx, __shfl_xor(mx, msk, 32));
      const float mnew = fmaxf(mrow[r], mx);
      const float sf = __expf(mrow[r] - mnew);
      const float e0 = __expf(x0 - mnew), e1 = __expf(x1 - mnew);
      float sum = e0 + e1;
#pragma unroll
      for (int msk = 1; msk <= 8; msk <<= 1) sum += __shfl_xor(sum, msk, 32);
      lrow[r] = lrow[r] * sf + sum;
      mrow[r] = mnew;
#pragma unroll
      for (int f = 0; f < 4; ++f) acc[f][r] *= sf;

      __bf16* prow = Ps + wave * 512 + (r + 8 * hi) * 32;
      prow[nl]      = (__bf16)e0;
      prow[nl + 16] = (__bf16)e1;
    }

    // ---- O += P @ V (A-frag from per-wave LDS bounce) ----
    {
      const v16bf pa = load_frag(Ps + wave * 512 + nl * 32, 0, ko);
#pragma unroll
      for (int f = 0; f < 4; ++f) {
        const v16bf vf = load_frag(Vs + (f * 16 + nl) * 32, 0, ko);
        acc[f] = wmma_bf16(pa, vf, acc[f]);
      }
    }
    __syncthreads();   // protect Ks/Vs (and staging reuse) for next iteration
  }

  // ---- normalize + write merged-head output ----
#pragma unroll
  for (int f = 0; f < 4; ++f) {
#pragma unroll
    for (int r = 0; r < 8; ++r) {
      const int s = qbase + r + 8 * hi;
      a_tmp[((size_t)b * SEQ + s) * NX + h * DH + f * 16 + nl] = acc[f][r] / lrow[r];
    }
  }
}

// ---------------------------------------------------------------------------
extern "C" void kernel_launch(void* const* d_in, const int* in_sizes, int n_in,
                              void* d_out, int out_size, void* d_ws, size_t ws_size,
                              hipStream_t stream)
{
  const float* x      = (const float*)d_in[0];
  const float* visf   = (const float*)d_in[1];
  const float* tagse  = (const float*)d_in[2];
  const float* W_attn = (const float*)d_in[3];
  const float* b_attn = (const float*)d_in[4];
  const float* W_vis  = (const float*)d_in[5];
  const float* b_vis  = (const float*)d_in[6];
  const float* W_tags = (const float*)d_in[7];
  const float* b_tags = (const float*)d_in[8];
  const float* W_proj = (const float*)d_in[9];
  const float* b_proj = (const float*)d_in[10];

  float* out     = (float*)d_out;
  float* a_out   = out;                               // [8,1024,1024]
  float* present = out + (size_t)8 * SEQ * NX;        // [8,2,16,1087,64]
  float* qtmp    = a_out;           // Q scratch aliases 'a' (proj overwrites last)
  float* a_tmp   = (float*)d_ws;    // 8192*1024 f32 = 33.5 MB of scratch

  const dim3 blk(128);

  // 1) qkv = x @ W_attn + b_attn  -> Q scratch + present text K/V (rows 63..1086)
  gemm_bf16_wmma<1, false><<<dim3(48, 128), blk, 0, stream>>>(
      x, 1024, W_attn, 3072, b_attn, qtmp, present, 8192, 3072, 1024, 0, 1);
  // 2) visual  -> present rows 14..62   (M=392 ragged)
  gemm_bf16_wmma<2, true><<<dim3(32, 7), blk, 0, stream>>>(
      visf, 1024, W_vis, 2048, b_vis, nullptr, present, 392, 2048, 1024, 14, 49);
  // 3) tags    -> present rows 0..13    (M=112, K=400 ragged)
  gemm_bf16_wmma<2, true><<<dim3(32, 2), blk, 0, stream>>>(
      tagse, 400, W_tags, 2048, b_tags, nullptr, present, 112, 2048, 400, 0, 14);
  // 4) fused attention -> a_tmp
  attn_fused<<<dim3(16, 16, 8), blk, 0, stream>>>(qtmp, present, a_tmp);
  // 5) a = a_tmp @ W_proj + b_proj
  gemm_bf16_wmma<0, false><<<dim3(16, 128), blk, 0, stream>>>(
      a_tmp, 1024, W_proj, 1024, b_proj, a_out, present, 8192, 1024, 1024, 0, 1);
}
